// SCRM_38680475468355
// MI455X (gfx1250) — matrix operand order repaired and checked by reading, not compiled, not run on hardware
//
#include <hip/hip_runtime.h>

// MI455X / gfx1250: wave32, WMMA bf16 16x16x32 path.
// Pipeline:
//   K1: elementwise (cfc combine + BN1d + LeakyReLU) -> bf16 zero-padded [32,128,58,58] in ws
//   K2: conv weights -> WMMA A-fragment packed bf16 in ws (L2-resident, 294 KB)
//   K3: implicit-GEMM 3x3 conv via v_wmma_f32_16x16x32_bf16 + bias + BN2d + LeakyReLU -> f32 d_out
//       Block tile widened to 32 output columns: 2 accumulators per wave, each A-fragment
//       feeds 2 WMMAs (halves A traffic, doubles matrix work per barrier).

#define B_     32
#define C_     128
#define H_     56
#define W_     56
#define HP     58
#define WP     58
#define SIZE_  (C_ * H_ * W_)     // 401408
#define KTOT   (C_ * 9)           // 1152
#define NKB    (KTOT / 32)        // 36 K-blocks of 32
#define EPS_   1e-5f
#define SLOPE_ 0.1f

typedef __attribute__((ext_vector_type(16))) __bf16 v16bf;
typedef __attribute__((ext_vector_type(8)))  float  v8f;

static constexpr size_t ZPAD_ELEMS  = (size_t)B_ * C_ * HP * WP;   // 13,778,944 bf16
static constexpr size_t WPACK_ELEMS = (size_t)8 * NKB * 32 * 16;   // 147,456 bf16
static constexpr size_t ZPAD_BYTES  = ZPAD_ELEMS * 2;              // 27,557,888 (32B-aligned)

// float -> bf16 round-to-nearest-even (raw bits)
__device__ __forceinline__ unsigned short f32_to_bf16(float f) {
  unsigned int u = __float_as_uint(f);
  unsigned int lsb = (u >> 16) & 1u;
  u += 0x7fffu + lsb;
  return (unsigned short)(u >> 16);
}

// ---------------- Kernel 1: fused elementwise stage + zero-halo pad ----------------
__global__ __launch_bounds__(256)
void fused_cfc_bn_lrelu_pad(const float* __restrict__ ax, const float* __restrict__ mx,
                            const float* __restrict__ cfc_w,
                            const float* __restrict__ bn_g, const float* __restrict__ bn_b,
                            const float* __restrict__ bn_m, const float* __restrict__ bn_v,
                            unsigned short* __restrict__ zpad) {
  size_t gid = (size_t)blockIdx.x * blockDim.x + threadIdx.x;
  if (gid >= ZPAD_ELEMS) return;
  int    xp = (int)(gid % WP);
  size_t r1 = gid / WP;
  int    yp = (int)(r1 % HP);
  size_t r2 = r1 / HP;
  int    c  = (int)(r2 % C_);
  int    b  = (int)(r2 / C_);
  float z = 0.0f;
  if (xp > 0 && xp < WP - 1 && yp > 0 && yp < HP - 1) {
    int    i  = (c * H_ + (yp - 1)) * W_ + (xp - 1);
    size_t gi = (size_t)b * SIZE_ + i;
    float a = ax[gi];
    float m = mx[gi];
    float zz = cfc_w[2 * i] * a + cfc_w[2 * i + 1] * m;
    float s  = bn_g[i] * rsqrtf(bn_v[i] + EPS_);
    zz = (zz - bn_m[i]) * s + bn_b[i];
    z = zz > 0.0f ? zz : SLOPE_ * zz;
  }
  zpad[gid] = f32_to_bf16(z);
}

// ---------------- Kernel 2: repack conv weights into WMMA A-fragment layout ----------------
// wpack[(((mt*NKB + kb)*32 + lane)*16 + j)] with contiguous 16-bit A 16x32 lane layout:
//   lane<16 : K = j        (K 0..15)
//   lane>=16: K = j+16     (K 16..31)
// K ordering: k = tap*128 + cin, tap = ky*3+kx  (each K=32 block lies within one tap)
__global__ __launch_bounds__(256)
void repack_weights(const float* __restrict__ w, unsigned short* __restrict__ wp) {
  int g = blockIdx.x * blockDim.x + threadIdx.x;
  if (g >= (int)WPACK_ELEMS) return;
  int j    = g & 15;
  int lane = (g >> 4) & 31;
  int kb   = (g >> 9) % NKB;
  int mt   = g / (NKB * 512);
  int koff = (lane >> 4) * 16 + j;
  int k    = kb * 32 + koff;
  int tap  = k >> 7;       // k / 128
  int cin  = k & 127;
  int cout = mt * 16 + (lane & 15);
  float v = w[(size_t)cout * (C_ * 9) + (size_t)cin * 9 + tap];
  wp[g] = f32_to_bf16(v);
}

// ---------------- Kernel 3: implicit-GEMM 3x3 conv via WMMA + BN2 + LeakyReLU ----------------
// Block: 256 thr = 8 waves; wave w owns cout tile [16w, 16w+16) over 32 output columns
// (two 16-wide N sub-tiles -> 2 accumulators, shared A-fragment).
// Block covers (b, y, 32 output x). K loop: 3 ky x 4 cin-blocks, strip reused over 3 kx.
__global__ __launch_bounds__(256)
void conv3x3_wmma_bn_lrelu(const unsigned short* __restrict__ zpad,
                           const unsigned short* __restrict__ wpack,
                           const float* __restrict__ conv_b,
                           const float* __restrict__ g2, const float* __restrict__ bt2,
                           const float* __restrict__ mu2, const float* __restrict__ var2,
                           float* __restrict__ out) {
  // strip[col][cin] : transposed so a B-fragment = 16 contiguous cin per lane
  __shared__ __attribute__((aligned(64))) unsigned short strip[34][32];

  const int tid  = (int)threadIdx.x;
  const int wave = tid >> 5;
  const int lane = tid & 31;
  const int x0   = (int)blockIdx.x * 32;
  const int y    = (int)blockIdx.y;
  const int b    = (int)blockIdx.z;

  v8f acc[2];
  acc[0] = (v8f){};
  acc[1] = (v8f){};

  for (int ky = 0; ky < 3; ++ky) {
    const int yp = y + ky;
    for (int cb = 0; cb < 4; ++cb) {
      // ---- stage 32 cin x 34 cols of padded bf16 input into LDS (transposed) ----
      const unsigned short* zb =
          zpad + ((size_t)(b * C_ + cb * 32) * HP + yp) * WP;
      for (int e = tid; e < 32 * 34; e += 256) {
        int cin = e / 34;
        int col = e - cin * 34;
        int xp  = x0 + col;
        unsigned short v = 0;
        if (xp < WP) v = zb[(size_t)cin * (HP * WP) + xp];
        strip[col][cin] = v;
      }
      __syncthreads();

      // prefetch next strip's cachelines while this one computes
      {
        int ncb = cb + 1, nky = ky;
        if (ncb == 4) { ncb = 0; nky = ky + 1; }
        if (nky < 3) {
          const unsigned short* pz =
              zpad + ((size_t)(b * C_ + ncb * 32) * HP + (y + nky)) * WP + x0;
          __builtin_prefetch(pz, 0, 3);
        }
      }

      // B-fragment bases: N = sub*16 + (lane&15), K-half by lane>=16
      const unsigned short* bsrc0 = &strip[(lane & 15) + 0][(lane >> 4) * 16];
      const unsigned short* bsrc1 = &strip[(lane & 15) + 16][(lane >> 4) * 16];

#pragma unroll
      for (int kx = 0; kx < 3; ++kx) {
        const int kb = (ky * 3 + kx) * 4 + cb;
        v16bf afrag = *(const v16bf*)(wpack +
                        ((size_t)(wave * NKB + kb) * 32 + lane) * 16);
        v16bf bf0 = *(const v16bf*)(bsrc0 + (size_t)kx * 32);
        v16bf bf1 = *(const v16bf*)(bsrc1 + (size_t)kx * 32);
        acc[0] = __builtin_amdgcn_wmma_f32_16x16x32_bf16(
                     false, afrag, false, bf0, (short)0, acc[0], false, false);
        acc[1] = __builtin_amdgcn_wmma_f32_16x16x32_bf16(
                     false, afrag, false, bf1, (short)0, acc[1], false, false);
      }
      __syncthreads();
    }
  }

  // ---- epilogue: bias + BN2d + LeakyReLU, f32 store ----
  const int n    = lane & 15;
  const int half = (lane >> 4) * 8;
#pragma unroll
  for (int sub = 0; sub < 2; ++sub) {
    const int x = x0 + sub * 16 + n;
    if (x < W_) {
#pragma unroll
      for (int r = 0; r < 8; ++r) {
        int cout = wave * 16 + r + half;
        float s = g2[cout] * rsqrtf(var2[cout] + EPS_);
        float v = (acc[sub][r] + conv_b[cout] - mu2[cout]) * s + bt2[cout];
        v = v > 0.0f ? v : SLOPE_ * v;
        out[(((size_t)b * C_ + cout) * H_ + y) * W_ + x] = v;
      }
    }
  }
}

// ---------------- driver ----------------
extern "C" void kernel_launch(void* const* d_in, const int* in_sizes, int n_in,
                              void* d_out, int out_size, void* d_ws, size_t ws_size,
                              hipStream_t stream) {
  (void)in_sizes; (void)n_in; (void)out_size; (void)ws_size;

  const float* ax     = (const float*)d_in[0];
  const float* mx     = (const float*)d_in[1];
  const float* cfc_w  = (const float*)d_in[2];
  const float* bn_g   = (const float*)d_in[3];
  const float* bn_b   = (const float*)d_in[4];
  const float* bn_m   = (const float*)d_in[5];
  const float* bn_v   = (const float*)d_in[6];
  const float* conv_w = (const float*)d_in[7];
  const float* conv_b = (const float*)d_in[8];
  const float* g2     = (const float*)d_in[9];
  const float* bt2    = (const float*)d_in[10];
  const float* mu2    = (const float*)d_in[11];
  const float* var2   = (const float*)d_in[12];

  unsigned short* zpad  = (unsigned short*)d_ws;
  unsigned short* wpack = (unsigned short*)((char*)d_ws + ZPAD_BYTES);
  float* out = (float*)d_out;

  {
    int blocks = (int)((ZPAD_ELEMS + 255) / 256);
    fused_cfc_bn_lrelu_pad<<<blocks, 256, 0, stream>>>(ax, mx, cfc_w, bn_g, bn_b,
                                                       bn_m, bn_v, zpad);
  }
  {
    int blocks = (int)((WPACK_ELEMS + 255) / 256);
    repack_weights<<<blocks, 256, 0, stream>>>(conv_w, wpack);
  }
  {
    dim3 grid(2, H_, B_);  // 32-px x-tiles (56 -> 2 tiles), 56 rows, 32 batches
    conv3x3_wmma_bn_lrelu<<<grid, 256, 0, stream>>>(zpad, wpack, conv_b,
                                                    g2, bt2, mu2, var2, out);
  }
}
